// GlobalAttentionGeneral_37984690766552
// MI455X (gfx1250) — matrix operand order repaired and checked by reading, not compiled, not run on hardware
//
#include <hip/hip_runtime.h>

// ---------- WMMA types / helpers (CDNA5 wave32, 16x16x32 bf16) ----------
typedef __bf16 bf16_t;
typedef __attribute__((ext_vector_type(4)))  __bf16 bf16x4;
typedef __attribute__((ext_vector_type(8)))  __bf16 bf16x8;
typedef __attribute__((ext_vector_type(16))) __bf16 bf16x16;
typedef __attribute__((ext_vector_type(8)))  float  f32x8;

__device__ __forceinline__ f32x8 wmma_bf16(bf16x16 a, bf16x16 b, f32x8 c) {
  return __builtin_amdgcn_wmma_f32_16x16x32_bf16(false, a, false, b, (short)0, c,
                                                 false, false);
}

// Fragment layout: each 16x16x32 operand tile = 512 bf16, [lane 32][slot 16].
// A (16Mx32K): lane = m + 16*((k>>4)&1), slot = k & 15
// B (32Kx16N): lane = n + 16*(k>>4),     slot = k & 15
__device__ __forceinline__ int a_idx(int m, int k) {
  return (((m & 15) + (((k >> 4) & 1) << 4)) << 4) | (k & 15);
}
__device__ __forceinline__ int b_idx(int k, int n) {
  return (((n & 15) + ((k >> 4) << 4)) << 4) | (k & 15);
}
__device__ __forceinline__ bf16x4 cvt4(float4 v) {
  bf16x4 r;
  r[0] = (bf16_t)v.x; r[1] = (bf16_t)v.y; r[2] = (bf16_t)v.z; r[3] = (bf16_t)v.w;
  return r;
}
__device__ __forceinline__ bf16x16 frag(const bf16_t* p) {
  return *(const bf16x16*)p;
}

#define NB   64
#define CLN  32
#define QL   32
#define IH   16
#define IW   16
#define CDF  256
#define SL   128
#define IDF  8192

// Scratch (bf16 elements)
#define WB_OFF   0           // Wb  : [kchunk 256][ntile 16][512]   2,097,152
#define WA_OFF   2097152     // Wa  : [mtile 512][kchunk 8][512]    2,097,152
#define CB_OFF   4194304     // ctxB: [b][kchunk 8][ntile 8][512]   2,097,152
#define CA_OFF   6291456     // ctxA: [b][mtile 16][kchunk 4][512]  2,097,152
#define T2_OFF   8388608     // T2a : [b][mtile 2][kchunk 8][512]     524,288
#define AT_OFF   8912896     // attnB:[b][kchunk 4][ntile 2][512]     262,144
#define RT_OFF   9175040     // Rtb : [b][kchunk 8][ntile 2][512]     524,288

// =====================================================================
// k0w: W[8192][256] -> Wb (B-frag, K=d,N=c) and Wa (A-frag, M=d,K=c).
// =====================================================================
__global__ __launch_bounds__(256) void k0w(const float* __restrict__ W,
                                           bf16_t* __restrict__ Wb,
                                           bf16_t* __restrict__ Wa) {
  __shared__ __align__(16) bf16_t La[8192];
  __shared__ __align__(16) bf16_t Lb[8192];
  const int d0 = blockIdx.x * 32;
  const int t = threadIdx.x;
#pragma unroll
  for (int j = 0; j < 8; ++j) {
    int f = t + j * 256;
    int dloc = f >> 6, c0 = (f & 63) * 4;
    float4 v = *(const float4*)(W + (size_t)(d0 + dloc) * CDF + c0);
    *(bf16x4*)(La + ((dloc >> 4) * 8 + (c0 >> 5)) * 512 + a_idx(dloc & 15, c0 & 31)) =
        cvt4(v);
    float vals[4] = {v.x, v.y, v.z, v.w};
    bf16_t* lb = Lb + (c0 >> 4) * 512;
#pragma unroll
    for (int u = 0; u < 4; ++u)
      lb[b_idx(dloc, (c0 + u) & 15)] = (bf16_t)vals[u];
  }
  __syncthreads();
  const int o = t * 32;  // contiguous copy-out: 8192 elems per region
#pragma unroll
  for (int u = 0; u < 4; ++u) {
    *(bf16x8*)(Wa + (size_t)d0 * 256 + o + u * 8) = *(const bf16x8*)(La + o + u * 8);
    *(bf16x8*)(Wb + (size_t)d0 * 256 + o + u * 8) = *(const bf16x8*)(Lb + o + u * 8);
  }
}

// =====================================================================
// k0c: context[b][256][128] -> ctxB (B-frag) and ctxA (A-frag).
// =====================================================================
__global__ __launch_bounds__(256) void k0c(const float* __restrict__ context,
                                           bf16_t* __restrict__ ctxB,
                                           bf16_t* __restrict__ ctxA) {
  __shared__ __align__(16) bf16_t La[8192];
  __shared__ __align__(16) bf16_t Lb[8192];
  const int b = blockIdx.x >> 2;
  const int c0 = (blockIdx.x & 3) * 64;
  const int t = threadIdx.x;
  const float* src = context + (size_t)b * (CDF * SL) + (size_t)c0 * SL;
#pragma unroll
  for (int j = 0; j < 8; ++j) {
    int f = t + j * 256;
    int cloc = f >> 5, s0 = (f & 31) * 4;
    float4 v = *(const float4*)(src + (size_t)cloc * SL + s0);
    *(bf16x4*)(La + ((cloc >> 4) * 4 + (s0 >> 5)) * 512 + a_idx(cloc & 15, s0 & 31)) =
        cvt4(v);
    float vals[4] = {v.x, v.y, v.z, v.w};
    bf16_t* lb = Lb + ((cloc >> 5) * 8 + (s0 >> 4)) * 512;
#pragma unroll
    for (int u = 0; u < 4; ++u)
      lb[b_idx(cloc & 31, (s0 + u) & 15)] = (bf16_t)vals[u];
  }
  __syncthreads();
  const size_t gbase = (size_t)b * 32768 + (size_t)c0 * 128;
  const int o = t * 32;
#pragma unroll
  for (int u = 0; u < 4; ++u) {
    *(bf16x8*)(ctxA + gbase + o + u * 8) = *(const bf16x8*)(La + o + u * 8);
    *(bf16x8*)(ctxB + gbase + o + u * 8) = *(const bf16x8*)(Lb + o + u * 8);
  }
}

// =====================================================================
// k1: T2[b][q][c'] = sum_d targetT[b][q][d] * W[d][c'].
//   grid (64,4). A staged via LDS (transpose), B-frags direct from Wb.
//   K unrolled x4: one barrier pair per 4 WMMA. Output emitted as bf16
//   A-fragment layout (T2a) through an LDS exchange.
// =====================================================================
__global__ __launch_bounds__(256) void k1_t2(const float* __restrict__ input,
                                             const bf16_t* __restrict__ Wb,
                                             bf16_t* __restrict__ T2a) {
  __shared__ __align__(32) bf16_t Af[4 * 1024];  // [chunk 4][mtile 2][512]
  __shared__ float Sf[QL * 64];                  // 32x64 result exchange
  const int b = blockIdx.x;
  const int n0 = blockIdx.y * 64;
  const int t = threadIdx.x;
  const int lane = t & 31;
  const int wid = t >> 5;
  const int m_t = wid & 1;
  const int n_t = wid >> 1;
  const float* inb = input + (size_t)b * (CLN * QL * 256);

  const int aq = t >> 3, adk = (t & 7) * 4;
  bf16_t* a_st0 = Af + (aq >> 4) * 512 + a_idx(aq & 15, adk);
  const bf16_t* afp0 = Af + m_t * 512 + lane * 16;
  const bf16_t* wbp = Wb + (size_t)(blockIdx.y * 4 + n_t) * 512 + lane * 16;

  f32x8 acc = {};
  for (int kk = 0; kk < IDF; kk += 128) {
    const int c = kk >> 8, hw = kk & 255;  // all 4 chunks share one (h,w) block
    const float* ap = inb + (c * QL + aq) * 256 + hw + adk;
#pragma unroll
    for (int ch = 0; ch < 4; ++ch) {
      float4 av = *(const float4*)(ap + ch * 32);
      *(bf16x4*)(a_st0 + ch * 1024) = cvt4(av);
    }
    __syncthreads();
    const int kc = kk >> 5;
#pragma unroll
    for (int ch = 0; ch < 4; ++ch)
      acc = wmma_bf16(frag(afp0 + ch * 1024),
                      frag(wbp + (size_t)(kc + ch) * 8192), acc);
    __syncthreads();
  }
  // exchange through LDS, emit T2a in A-frag layout (contiguous b128 stores)
  const int n = n_t * 16 + (lane & 15);
  const int mbase = m_t * 16 + ((lane >> 4) << 3);
#pragma unroll
  for (int r = 0; r < 8; ++r) Sf[(mbase + r) * 64 + n] = acc[r];
  __syncthreads();
  {
    const int p = t * 8;
    const int lt = p >> 9;           // [mtile_loc 2][kchunk_loc 2]
    const int o = p & 511;
    const int mt_l = lt >> 1, kc_l = lt & 1;
    const int lane_a = o >> 4, slot0 = o & 15;
    const int m = mt_l * 16 + (lane_a & 15);
    const int kbase = ((lane_a >> 4) << 4) + slot0;
    const int cl = kc_l * 32 + kbase;  // 8 consecutive local c'
    bf16x8 pk;
#pragma unroll
    for (int u = 0; u < 8; ++u) pk[u] = (bf16_t)Sf[m * 64 + cl + u];
    const int gt = mt_l * 8 + (n0 >> 5) + kc_l;  // [mtile 2][kchunk 8]
    *(bf16x8*)(T2a + (size_t)b * 8192 + gt * 512 + o) = pk;
  }
}

// =====================================================================
// k2: attn = softmax(T2 x context); barrier-free fragment K-loop.
//   Writes attnT (f32 output) and attnB (bf16 B-frag for k3).
// =====================================================================
__global__ __launch_bounds__(256) void k2_attn(const bf16_t* __restrict__ T2a,
                                               const bf16_t* __restrict__ ctxB,
                                               bf16_t* __restrict__ attnB,
                                               float* __restrict__ attnT) {
  __shared__ float Sm[QL * SL];
  const int b = blockIdx.x;
  const int t = threadIdx.x;
  const int lane = t & 31;
  const int wid = t >> 5;  // N-tile

  const bf16_t* t2p = T2a + (size_t)b * 8192 + lane * 16;
  const bf16_t* cbp = ctxB + (size_t)b * 32768 + wid * 512 + lane * 16;

  f32x8 acc0 = {}, acc1 = {};
#pragma unroll
  for (int kc = 0; kc < 8; ++kc) {
    bf16x16 bfr = frag(cbp + (size_t)kc * 4096);
    acc0 = wmma_bf16(frag(t2p + (size_t)kc * 512), bfr, acc0);
    acc1 = wmma_bf16(frag(t2p + 4096 + (size_t)kc * 512), bfr, acc1);
  }
  const int n = wid * 16 + (lane & 15);
  const int h8 = (lane >> 4) << 3;
#pragma unroll
  for (int r = 0; r < 8; ++r) {
    Sm[(h8 + r) * SL + n] = acc0[r];
    Sm[(16 + h8 + r) * SL + n] = acc1[r];
  }
  __syncthreads();
  if (t < 32) {  // softmax row q=t
    float mx = -3.4e38f;
    for (int s = 0; s < SL; ++s) mx = fmaxf(mx, Sm[t * SL + s]);
    float sum = 0.f;
    for (int s = 0; s < SL; ++s) {
      float e = __expf(Sm[t * SL + s] - mx);
      Sm[t * SL + s] = e;
      sum += e;
    }
    float inv = 1.f / sum;
    bf16_t* ab = attnB + (size_t)b * 4096;
    float* at = attnT + (size_t)b * (SL * QL);
    for (int s0 = 0; s0 < SL; s0 += 4) {
      bf16x4 pk;
#pragma unroll
      for (int u = 0; u < 4; ++u) {
        float v = Sm[t * SL + s0 + u] * inv;
        at[(s0 + u) * QL + t] = v;
        pk[u] = (bf16_t)v;
      }
      int lane_b = (t & 15) + (((s0 >> 4) & 1) << 4);
      *(bf16x4*)(ab + ((s0 >> 5) * 2 + (t >> 4)) * 512 + lane_b * 16 + (s0 & 15)) = pk;
    }
  }
}

// =====================================================================
// k3: R = context x attn^T; pure fragment GEMM (no LDS/barriers).
//   Output Rtb in B-frag layout for k4 (8-elem runs -> one b128).
// =====================================================================
__global__ __launch_bounds__(256) void k3_r(const bf16_t* __restrict__ ctxA,
                                            const bf16_t* __restrict__ attnB,
                                            bf16_t* __restrict__ Rtb) {
  const int b = blockIdx.x;
  const int t = threadIdx.x;
  const int lane = t & 31;
  const int wid = t >> 5;  // M-tiles {2w,2w+1}, N-tiles {0,1}

  const bf16_t* cap = ctxA + (size_t)b * 32768 + lane * 16;
  const bf16_t* abp = attnB + (size_t)b * 4096 + lane * 16;

  f32x8 acc[2][2] = {{{}, {}}, {{}, {}}};
#pragma unroll
  for (int kc = 0; kc < 4; ++kc) {
    bf16x16 b0 = frag(abp + (size_t)kc * 1024);
    bf16x16 b1 = frag(abp + (size_t)kc * 1024 + 512);
#pragma unroll
    for (int mt = 0; mt < 2; ++mt) {
      bf16x16 af = frag(cap + (size_t)((2 * wid + mt) * 4 + kc) * 512);
      acc[mt][0] = wmma_bf16(af, b0, acc[mt][0]);
      acc[mt][1] = wmma_bf16(af, b1, acc[mt][1]);
    }
  }
  const int h8 = (lane >> 4) << 3;
  bf16_t* rb = Rtb + (size_t)b * 8192;
#pragma unroll
  for (int mt = 0; mt < 2; ++mt)
#pragma unroll
    for (int nt = 0; nt < 2; ++nt) {
      int mb = (2 * wid + mt) * 16 + h8;  // 8 consecutive c (K of k4)
      int q = nt * 16 + (lane & 15);
      bf16x8 pk;
#pragma unroll
      for (int r = 0; r < 8; ++r) pk[r] = (bf16_t)acc[mt][nt][r];
      int lane_b = (q & 15) + (((mb >> 4) & 1) << 4);
      *(bf16x8*)(rb + ((mb >> 5) * 2 + nt) * 512 + lane_b * 16 + (mb & 15)) = pk;
    }
}

// =====================================================================
// k4: wc = W x R with fused .view+transpose store; pure fragment GEMM.
// =====================================================================
__global__ __launch_bounds__(256) void k4_wc(const bf16_t* __restrict__ Wa,
                                             const bf16_t* __restrict__ Rtb,
                                             float* __restrict__ out) {
  const int b = blockIdx.x;
  const int t = threadIdx.x;
  const int lane = t & 31;
  const int wid = t >> 5;
  const int m0 = blockIdx.y * 256;

  const bf16_t* wap = Wa + (size_t)(blockIdx.y * 16) * 4096 + lane * 16;
  const bf16_t* rbp = Rtb + (size_t)b * 8192 + lane * 16;

  f32x8 acc[2][2] = {{{}, {}}, {{}, {}}};
#pragma unroll
  for (int kc = 0; kc < 8; ++kc) {
    bf16x16 b0 = frag(rbp + (size_t)kc * 1024);
    bf16x16 b1 = frag(rbp + (size_t)kc * 1024 + 512);
#pragma unroll
    for (int mt = 0; mt < 2; ++mt) {
      bf16x16 af = frag(wap + (size_t)((2 * wid + mt) * 8 + kc) * 512);
      acc[mt][0] = wmma_bf16(af, b0, acc[mt][0]);
      acc[mt][1] = wmma_bf16(af, b1, acc[mt][1]);
    }
  }
  // Permuted store: flat j = d*32 + q; out[b][c][q'][h][w] with
  // c=(d>>3)&31, q'=d>>8, h=(2d+(q>>4))&15, w=q&15.
  const int h8 = (lane >> 4) << 3;
  const size_t ob = (size_t)b * (CLN * QL * IH * IW);
#pragma unroll
  for (int mt = 0; mt < 2; ++mt)
#pragma unroll
    for (int nt = 0; nt < 2; ++nt)
#pragma unroll
      for (int r = 0; r < 8; ++r) {
        int d = m0 + (2 * wid + mt) * 16 + h8 + r;
        int w = lane & 15;
        int h = (2 * d + nt) & 15;
        size_t idx = ob + (size_t)((d >> 3) & 31) * 8192 +
                     (size_t)(d >> 8) * 256 + h * 16 + w;
        out[idx] = acc[mt][nt][r];
      }
}

extern "C" void kernel_launch(void* const* d_in, const int* in_sizes, int n_in,
                              void* d_out, int out_size, void* d_ws, size_t ws_size,
                              hipStream_t stream) {
  const float* input   = (const float*)d_in[0];  // [64,32,32,16,16]
  const float* context = (const float*)d_in[1];  // [64,256,128]
  const float* W       = (const float*)d_in[2];  // [8192,256]

  float* out_wc    = (float*)d_out;
  float* out_attnT = out_wc + (size_t)NB * CLN * QL * IH * IW;

  bf16_t* ws   = (bf16_t*)d_ws;
  bf16_t* Wb   = ws + WB_OFF;
  bf16_t* Wa   = ws + WA_OFF;
  bf16_t* ctxB = ws + CB_OFF;
  bf16_t* ctxA = ws + CA_OFF;
  bf16_t* T2a  = ws + T2_OFF;
  bf16_t* attB = ws + AT_OFF;
  bf16_t* Rtb  = ws + RT_OFF;

  k0w    <<<dim3(256),    256, 0, stream>>>(W, Wb, Wa);
  k0c    <<<dim3(256),    256, 0, stream>>>(context, ctxB, ctxA);
  k1_t2  <<<dim3(NB, 4),  256, 0, stream>>>(input, Wb, T2a);
  k2_attn<<<dim3(NB),     256, 0, stream>>>(T2a, ctxB, attB, out_attnT);
  k3_r   <<<dim3(NB),     256, 0, stream>>>(ctxA, attB, Rtb);
  k4_wc  <<<dim3(NB, 32), 256, 0, stream>>>(Wa, Rtb, out_wc);
}
